// AttentionOCR_80779744903855
// MI455X (gfx1250) — compile-verified, matrix operand order checked by reading, not run on hardware
//
#include <hip/hip_runtime.h>

#define B_    512
#define P_    64
#define CIN_  512
#define HID_  512
#define EMB_  512
#define NCLS_ 96
#define T_    30

typedef __attribute__((ext_vector_type(8)))  float   v8f;    // WMMA accumulator
typedef __attribute__((ext_vector_type(8)))  float   v8fv;   // staging vector
typedef __attribute__((ext_vector_type(8)))  __bf16  v8bf;
typedef __attribute__((ext_vector_type(16))) __bf16  v16bf;

__device__ __forceinline__ unsigned short f2bf(float f) {
  union { __bf16 h; unsigned short u; } x;
  x.h = (__bf16)f;
  return x.u;
}
__device__ __forceinline__ float sigf(float x) { return 1.0f / (1.0f + __expf(-x)); }

union Frag32 { v16bf v; uint4 q[2]; };
union Bf8Pack { v8bf b; uint4 q; };

// packed float8 -> bf16x8 (4x v_cvt_pk_bf16_f32, contiguous dest regs)
__device__ __forceinline__ uint4 cvt8(const float* __restrict__ src) {
  v8fv f = *(const v8fv*)src;          // 2x global_load_b128 (64B-aligned call sites)
  Bf8Pack p;
  p.b = __builtin_convertvector(f, v8bf);
  return p.q;
}

// -------------------------------------------------------------------------
// bf16 WMMA GEMM:  C[M,N] = concatK(A0,A1) @ concatK(W0,W1)^T + b0 + b1
//   AT = unsigned short (bf16) or float (converted during staging)
//   W parts are pre-converted bf16 [N,K] row-major.
//   Requires: K0 % 32 == 0, M % 128 == 0 (true for all call sites).
//   Block tile 128x64, 8 waves of 32x32, K-step 32.
// -------------------------------------------------------------------------
#define BM 128
#define BN 64
#define BK 32

template <typename AT>
__global__ __launch_bounds__(256)
void gemm_wmma(const AT* __restrict__ a0, int lda0,
               const AT* __restrict__ a1, int lda1,
               int K0, int Ktot,
               const unsigned short* __restrict__ w0, int ldw0,
               const unsigned short* __restrict__ w1, int ldw1,
               const float* __restrict__ bias0,
               const float* __restrict__ bias1,
               float* __restrict__ Cf,
               unsigned short* __restrict__ Cbf,
               int ldc, int M, int N)
{
  __shared__ __align__(16) unsigned short As[BM][BK];
  __shared__ __align__(16) unsigned short Bs[BN][BK];

  const int tid    = threadIdx.x;
  const int lane   = tid & 31;
  const int wave   = tid >> 5;
  const int wm     = wave & 3;     // 4 wave-rows  -> 128 M
  const int wn     = wave >> 2;    // 2 wave-cols  -> 64  N
  const int hs     = lane >> 4;    // K half-select (ISA 16-bit A/B layout)
  const int l16    = lane & 15;    // M (A) / N (B) index within 16
  const int blockM = blockIdx.y * BM;
  const int blockN = blockIdx.x * BN;

  // staging roles (fixed per thread)
  const int arow = tid >> 1, ahalf = tid & 1;   // A: 16 elems each
  const int brow = tid >> 2, bq    = tid & 3;   // W: 8  elems each
  const int nB   = blockN + brow;
  const bool wok = (nB < N);

  v8f acc[2][2] = {};

  // two sequential K segments -> loop-invariant base pointers, no per-iter selects
  #pragma unroll 1
  for (int part = 0; part < 2; ++part) {
    const AT*            ap  = part ? a1 : a0;
    const int            lda = part ? lda1 : lda0;
    const unsigned short* wp = part ? w1 : w0;
    const int            ldw = part ? ldw1 : ldw0;
    const int            Kp  = part ? (Ktot - K0) : K0;

    const AT*             abase = ap + (size_t)(blockM + arow) * lda + ahalf * 16;
    const unsigned short* wbase = wp + (size_t)nB * ldw + bq * 8;

    for (int kk = 0; kk < Kp; kk += BK) {
      // ---- stage A tile (128 x 32) ----
      if constexpr (sizeof(AT) == 2) {
        const unsigned short* src = (const unsigned short*)abase + kk;
        uint4 d0 = *(const uint4*)(src);
        uint4 d1 = *(const uint4*)(src + 8);
        if (kk + BK < Kp) __builtin_prefetch(src + BK, 0, 1);
        *(uint4*)&As[arow][ahalf * 16]     = d0;
        *(uint4*)&As[arow][ahalf * 16 + 8] = d1;
      } else {
        const float* srcf = (const float*)abase + kk;
        uint4 d0 = cvt8(srcf);
        uint4 d1 = cvt8(srcf + 8);
        if (kk + BK < Kp) __builtin_prefetch(srcf + BK, 0, 1);
        *(uint4*)&As[arow][ahalf * 16]     = d0;
        *(uint4*)&As[arow][ahalf * 16 + 8] = d1;
      }

      // ---- stage W tile (64 x 32) bf16 ----
      {
        uint4 d; d.x = d.y = d.z = d.w = 0u;
        if (wok) {
          const unsigned short* src = wbase + kk;
          d = *(const uint4*)src;
          if (kk + BK < Kp) __builtin_prefetch(src + BK, 0, 1);
        }
        *(uint4*)&Bs[brow][bq * 8] = d;
      }
      __syncthreads();

      // ---- fragments: two contiguous 8-elem K runs per lane -> 2x ds_load_b128 ----
      Frag32 fa[2], fb[2];
      #pragma unroll
      for (int ms = 0; ms < 2; ++ms) {
        const int r = (wm << 5) + (ms << 4) + l16;
        fa[ms].q[0] = *(const uint4*)&As[r][hs * 8];
        fa[ms].q[1] = *(const uint4*)&As[r][16 + hs * 8];
      }
      #pragma unroll
      for (int ns = 0; ns < 2; ++ns) {
        const int r = (wn << 5) + (ns << 4) + l16;
        fb[ns].q[0] = *(const uint4*)&Bs[r][hs * 8];
        fb[ns].q[1] = *(const uint4*)&Bs[r][16 + hs * 8];
      }

      #pragma unroll
      for (int ms = 0; ms < 2; ++ms)
        #pragma unroll
        for (int ns = 0; ns < 2; ++ns)
          acc[ms][ns] = __builtin_amdgcn_wmma_f32_16x16x32_bf16(
              false, fa[ms].v, false, fb[ns].v,
              (short)0, acc[ms][ns], false, false);
      __syncthreads();
    }
  }

  // ---- store: lane col = l16, VGPR v -> row v + 8*hs ----
  #pragma unroll
  for (int ns = 0; ns < 2; ++ns) {
    const int col = blockN + (wn << 5) + (ns << 4) + l16;
    if (col >= N) continue;
    float bv = 0.0f;
    if (bias0) bv += bias0[col];
    if (bias1) bv += bias1[col];
    #pragma unroll
    for (int ms = 0; ms < 2; ++ms) {
      const int rowBase = blockM + (wm << 5) + (ms << 4) + hs * 8;
      #pragma unroll
      for (int v = 0; v < 8; ++v) {
        const float val = acc[ms][ns][v] + bv;
        const size_t off = (size_t)(rowBase + v) * ldc + col;
        if (Cf)  Cf[off]  = val;
        if (Cbf) Cbf[off] = f2bf(val);
      }
    }
  }
}

// -------------------------------------------------------------------------
__global__ __launch_bounds__(256)
void cvt_f32_bf16(const float* __restrict__ s, unsigned short* __restrict__ d, int n) {
  int i = (blockIdx.x * 256 + threadIdx.x) * 8;
  if (i >= n) return;
  uint4 p = cvt8(s + i);
  *(uint4*)(d + i) = p;
}

__global__ __launch_bounds__(256)
void zero_u32(unsigned int* __restrict__ p, int n) {
  int i = blockIdx.x * 256 + threadIdx.x;
  if (i < n) p[i] = 0u;
}

__global__ __launch_bounds__(256)
void embed_gather(const float* __restrict__ tab, const int* __restrict__ tgt,
                  unsigned short* __restrict__ emb_bf, int t) {
  int idx = blockIdx.x * 256 + threadIdx.x;
  if (idx >= B_ * EMB_) return;
  int b  = idx >> 9;
  int e  = idx & (EMB_ - 1);
  int id = (t == 0) ? 0 : tgt[b * T_ + (t - 1)];
  emb_bf[idx] = f2bf(tab[(size_t)id * EMB_ + e]);
}

// softmax over 512 cols, one block per row (in place, fp32)
__global__ __launch_bounds__(256)
void softmax512_rows(float* __restrict__ x) {
  __shared__ float red[256];
  const int row = blockIdx.x, tid = threadIdx.x;
  float* p = x + (size_t)row * 512;
  float a0 = p[tid], a1 = p[tid + 256];
  red[tid] = fmaxf(a0, a1);
  __syncthreads();
  for (int s = 128; s > 0; s >>= 1) {
    if (tid < s) red[tid] = fmaxf(red[tid], red[tid + s]);
    __syncthreads();
  }
  const float m = red[0];
  __syncthreads();
  float e0 = __expf(a0 - m), e1 = __expf(a1 - m);
  red[tid] = e0 + e1;
  __syncthreads();
  for (int s = 128; s > 0; s >>= 1) {
    if (tid < s) red[tid] += red[tid + s];
    __syncthreads();
  }
  const float inv = 1.0f / red[0];
  p[tid] = e0 * inv;
  p[tid + 256] = e1 * inv;
}

// fused attention: scores = feats[b]·a[b], softmax over P, ctx = w·feats[b]
// one block per batch; feats (67 MB fp32) stays resident in the 192 MB L2
__global__ __launch_bounds__(256)
void attend(const float* __restrict__ feats, const float* __restrict__ a,
            unsigned short* __restrict__ ctx_bf) {
  __shared__ float sw[P_];
  const int b   = blockIdx.x;
  const int tid = threadIdx.x;
  const float* fb = feats + (size_t)b * P_ * HID_;
  const float* ab = a + (size_t)b * HID_;

  const int p = tid >> 2, sub = tid & 3;
  {
    const float* fp = fb + (size_t)p * HID_;
    float s = 0.0f;
    for (int h2 = sub; h2 < HID_; h2 += 4) s += fp[h2] * ab[h2];
    s += __shfl_xor(s, 1, 32);
    s += __shfl_xor(s, 2, 32);
    if (sub == 0) sw[p] = s;
  }
  __syncthreads();

  float mx = -1e30f;
  for (int i = 0; i < P_; ++i) mx = fmaxf(mx, sw[i]);
  float sum = 0.0f;
  for (int i = 0; i < P_; ++i) sum += __expf(sw[i] - mx);
  const float inv = 1.0f / sum;
  __syncthreads();
  if (tid < P_) sw[tid] = __expf(sw[tid] - mx) * inv;
  __syncthreads();

  for (int h2 = tid; h2 < HID_; h2 += 256) {
    float accv = 0.0f;
    #pragma unroll 4
    for (int pp = 0; pp < P_; ++pp) accv += sw[pp] * fb[(size_t)pp * HID_ + h2];
    ctx_bf[(size_t)b * HID_ + h2] = f2bf(accv);
  }
}

// LSTM elementwise (gate order i,f,g,o); c fp32 in place, h written as bf16
__global__ __launch_bounds__(256)
void lstm_update(const float* __restrict__ gates, float* __restrict__ c,
                 unsigned short* __restrict__ h_bf) {
  int idx = blockIdx.x * 256 + threadIdx.x;
  if (idx >= B_ * HID_) return;
  int b = idx >> 9;
  int n = idx & (HID_ - 1);
  const float* g = gates + (size_t)b * (4 * HID_);
  float gi = sigf(g[n]);
  float gf = sigf(g[n + HID_]);
  float gg = tanhf(g[n + 2 * HID_]);
  float go = sigf(g[n + 3 * HID_]);
  float cn = gf * c[idx] + gi * gg;
  c[idx] = cn;
  h_bf[idx] = f2bf(go * tanhf(cn));
}

// -------------------------------------------------------------------------
extern "C" void kernel_launch(void* const* d_in, const int* in_sizes, int n_in,
                              void* d_out, int out_size, void* d_ws, size_t ws_size,
                              hipStream_t stream) {
  (void)in_sizes; (void)n_in; (void)out_size; (void)ws_size;

  const float* features  = (const float*)d_in[0];
  const int*   targets   = (const int*)  d_in[1];
  const float* Wfc       = (const float*)d_in[3];
  const float* bfc       = (const float*)d_in[4];
  const float* emb_table = (const float*)d_in[5];
  const float* Wa        = (const float*)d_in[6];
  const float* ba        = (const float*)d_in[7];
  const float* Wc        = (const float*)d_in[8];
  const float* bc        = (const float*)d_in[9];
  const float* Wih       = (const float*)d_in[10];
  const float* Whh       = (const float*)d_in[11];
  const float* bih       = (const float*)d_in[12];
  const float* bhh       = (const float*)d_in[13];
  const float* Wo        = (const float*)d_in[14];
  const float* bo        = (const float*)d_in[15];
  float* out = (float*)d_out;

  // ---- workspace carve (256B aligned) ----
  char* wsp = (char*)d_ws;
  auto carve = [&](size_t bytes) -> void* {
    void* p = (void*)wsp;
    wsp += (bytes + 255) & ~(size_t)255;
    return p;
  };
  float* feats = (float*)carve((size_t)B_ * P_ * HID_ * 4);   // 67 MB
  float* c     = (float*)carve((size_t)B_ * HID_ * 4);
  float* alog  = (float*)carve((size_t)B_ * HID_ * 4);
  float* gates = (float*)carve((size_t)B_ * 4 * HID_ * 4);
  unsigned short* h_bf   = (unsigned short*)carve((size_t)B_ * HID_ * 2);
  unsigned short* emb_bf = (unsigned short*)carve((size_t)B_ * EMB_ * 2);
  unsigned short* ctx_bf = (unsigned short*)carve((size_t)B_ * HID_ * 2);
  unsigned short* x_bf   = (unsigned short*)carve((size_t)B_ * HID_ * 2);
  unsigned short* wfc_bf = (unsigned short*)carve((size_t)HID_ * CIN_ * 2);
  unsigned short* wa_bf  = (unsigned short*)carve((size_t)HID_ * (HID_ + EMB_) * 2);
  unsigned short* wc_bf  = (unsigned short*)carve((size_t)HID_ * (HID_ + EMB_) * 2);
  unsigned short* wih_bf = (unsigned short*)carve((size_t)4 * HID_ * HID_ * 2);
  unsigned short* whh_bf = (unsigned short*)carve((size_t)4 * HID_ * HID_ * 2);
  unsigned short* wo_bf  = (unsigned short*)carve((size_t)NCLS_ * HID_ * 2);

  dim3 blk(256);
  auto cvt = [&](const float* s, unsigned short* d, int n) {
    cvt_f32_bf16<<<(n / 8 + 255) / 256, blk, 0, stream>>>(s, d, n);
  };

  // ---- one-time weight conversion to bf16 (reused 30x) ----
  cvt(Wfc, wfc_bf, HID_ * CIN_);
  cvt(Wa,  wa_bf,  HID_ * (HID_ + EMB_));
  cvt(Wc,  wc_bf,  HID_ * (HID_ + EMB_));
  cvt(Wih, wih_bf, 4 * HID_ * HID_);
  cvt(Whh, whh_bf, 4 * HID_ * HID_);
  cvt(Wo,  wo_bf,  NCLS_ * HID_);

  // feats = features @ Wfc^T + bfc   (fp32 A path; M=32768)
  gemm_wmma<float><<<dim3(HID_ / BN, (B_ * P_) / BM), blk, 0, stream>>>(
      features, CIN_, features, CIN_, CIN_, CIN_,
      wfc_bf, CIN_, wfc_bf, CIN_, bfc, nullptr,
      feats, nullptr, HID_, B_ * P_, HID_);

  // h0 = c0 = 0
  zero_u32<<<(B_ * HID_ + 255) / 256, blk, 0, stream>>>((unsigned int*)c, B_ * HID_);
  zero_u32<<<(B_ * HID_ / 2 + 255) / 256, blk, 0, stream>>>((unsigned int*)h_bf, B_ * HID_ / 2);

  for (int t = 0; t < T_; ++t) {
    embed_gather<<<(B_ * EMB_) / 256, blk, 0, stream>>>(emb_table, targets, emb_bf, t);

    // a_logits = [h | emb] @ Wa^T + ba
    gemm_wmma<unsigned short><<<dim3(HID_ / BN, B_ / BM), blk, 0, stream>>>(
        h_bf, HID_, emb_bf, EMB_, HID_, HID_ + EMB_,
        wa_bf, HID_ + EMB_, wa_bf + HID_, HID_ + EMB_, ba, nullptr,
        alog, nullptr, HID_, B_, HID_);
    softmax512_rows<<<B_, blk, 0, stream>>>(alog);

    attend<<<B_, blk, 0, stream>>>(feats, alog, ctx_bf);

    // x = [emb | ctx] @ Wc^T + bc   (bf16 output straight into next GEMM's A)
    gemm_wmma<unsigned short><<<dim3(HID_ / BN, B_ / BM), blk, 0, stream>>>(
        emb_bf, EMB_, ctx_bf, HID_, EMB_, EMB_ + HID_,
        wc_bf, EMB_ + HID_, wc_bf + EMB_, EMB_ + HID_, bc, nullptr,
        nullptr, x_bf, HID_, B_, HID_);

    // gates = [x | h] @ [Wih | Whh]^T + bih + bhh
    gemm_wmma<unsigned short><<<dim3((4 * HID_) / BN, B_ / BM), blk, 0, stream>>>(
        x_bf, HID_, h_bf, HID_, HID_, 2 * HID_,
        wih_bf, HID_, whh_bf, HID_, bih, bhh,
        gates, nullptr, 4 * HID_, B_, 4 * HID_);

    lstm_update<<<(B_ * HID_) / 256, blk, 0, stream>>>(gates, c, h_bf);

    // out[:, t, :] = h @ Wo^T + bo
    gemm_wmma<unsigned short><<<dim3((NCLS_ + BN - 1) / BN, B_ / BM), blk, 0, stream>>>(
        h_bf, HID_, h_bf, HID_, HID_, HID_,
        wo_bf, HID_, wo_bf, HID_, bo, nullptr,
        out + (size_t)t * NCLS_, nullptr, T_ * NCLS_, B_, NCLS_);
  }
}
